// SemSegHead_31370441130276
// MI455X (gfx1250) — matrix operand order repaired and checked by reading, not compiled, not run on hardware
//
#include <hip/hip_runtime.h>
#include <math.h>

typedef __attribute__((ext_vector_type(16))) _Float16 v16h;
typedef __attribute__((ext_vector_type(8)))  _Float16 v8h;
typedef __attribute__((ext_vector_type(8)))  float    v8f;
typedef __attribute__((ext_vector_type(4)))  float    v4f;

// ---------------- workspace layout (byte offsets, all 256-aligned) ----------
constexpr size_t OFF_BT1   = 0;          // 256x256 f16   (up1 weights, BT [n][k])
constexpr size_t OFF_B1    = 131072;     // 256 f32
constexpr size_t OFF_BT2   = 132096;     // 128x64 f16
constexpr size_t OFF_B2    = 148480;     // 128 f32
constexpr size_t OFF_BTQKV = 148992;     // 48x32 f16
constexpr size_t OFF_BQKV  = 152064;     // 48 f32
constexpr size_t OFF_BTL1  = 152320;     // 32x288 f16
constexpr size_t OFF_BTL2  = 170752;     // 32x32 f16
constexpr size_t OFF_BTOC  = 172800;     // 32x32 f16
constexpr size_t OFF_FOLD  = 174848;     // 192 f32 (l1 s/b, l2 s/b, oc s/b)
constexpr size_t OFF_KSUM  = 175616;     // 16x8 f32
constexpr size_t OFF_VSUM  = 176128;     // 16x32 f32
constexpr size_t OFF_MAT   = 178176;     // 16x8x32 f32
constexpr size_t OFF_BTMAT = 194560;     // 16x48x32 f16
constexpr size_t OFF_HYP   = 243712;     // 16x32x32 f16 (hyperT, rows>=20 zero)
// big buffers
constexpr size_t OFF_U16   = 1048576;    // 16*16384*64 f16 (33.5MB) dead after k2
constexpr size_t OFF_QN    = 1048576;    // 16*65536*8 f16  (reuses U16)
constexpr size_t OFF_KN    = 17825792;   // 16*65536*8 f16
constexpr size_t OFF_UP0   = 34603008;   // 16*65536*32 f16 (67MB)
constexpr size_t OFF_V16   = 101711872;  // 16*65536*32 f16, dead after k3b
constexpr size_t OFF_PART  = 101711872;  // reuses V16
constexpr size_t OFF_UPF   = 168820736;  // 16*65536*32 f16
constexpr size_t MASK_ELEMS = (size_t)16 * 20 * 65536;

// ---------------- helpers ---------------------------------------------------
__device__ __forceinline__ v8f zero8() {
  v8f z;
#pragma unroll
  for (int i = 0; i < 8; ++i) z[i] = 0.f;
  return z;
}
__device__ __forceinline__ v16h zero16h() {
  v16h z;
#pragma unroll
  for (int i = 0; i < 16; ++i) z[i] = (_Float16)0;
  return z;
}
__device__ __forceinline__ v8f wmma16(v16h a, v16h b, v8f c) {
  return __builtin_amdgcn_wmma_f32_16x16x32_f16(false, a, false, b, (short)0, c, false, false);
}
// A fragment (16x32 f16): lane holds row lane&15; K halves depend on lane&16.
__device__ __forceinline__ v16h ldA16(const _Float16* row, int lane) {
  const int b0 = (lane & 16) ? 8 : 0;
  v8h lo = *(const v8h*)(row + b0);
  v8h hi = *(const v8h*)(row + 16 + b0);
  v16h a;
#pragma unroll
  for (int i = 0; i < 8; ++i) { a[i] = lo[i]; a[8 + i] = hi[i]; }
  return a;
}
// A fragment from f32 source
__device__ __forceinline__ v16h ldA32f(const float* row, int lane) {
  const int b0 = (lane & 16) ? 8 : 0;
  v4f x0 = *(const v4f*)(row + b0);
  v4f x1 = *(const v4f*)(row + b0 + 4);
  v4f x2 = *(const v4f*)(row + 16 + b0);
  v4f x3 = *(const v4f*)(row + 16 + b0 + 4);
  v16h a;
#pragma unroll
  for (int i = 0; i < 4; ++i) {
    a[i] = (_Float16)x0[i]; a[4 + i] = (_Float16)x1[i];
    a[8 + i] = (_Float16)x2[i]; a[12 + i] = (_Float16)x3[i];
  }
  return a;
}
// B fragment (32x16 f16) from BT layout [n][K] (per-column contiguous K)
__device__ __forceinline__ v16h ldB16(const _Float16* bt, int ld, int lane, int nt, int kofs) {
  const _Float16* p = bt + (size_t)(nt * 16 + (lane & 15)) * ld + kofs + ((lane & 16) ? 8 : 0);
  v8h lo = *(const v8h*)p;
  v8h hi = *(const v8h*)(p + 16);
  v16h b;
#pragma unroll
  for (int i = 0; i < 8; ++i) { b[i] = lo[i]; b[8 + i] = hi[i]; }
  return b;
}
__device__ __forceinline__ float gelu_exact(float x) {
  return 0.5f * x * (1.0f + erff(x * 0.70710678118f));
}

// ---------------- k0: weight pack / BN fold / accumulator zero --------------
__global__ void k0_pack(
    const float* up1_w, const float* up1_b, const float* up2_w, const float* up2_b,
    const float* q_w, const float* q_b, const float* k_w, const float* k_b,
    const float* v_w, const float* v_b, const float* l1_w, const float* l1_b,
    const float* l2_w, const float* l2_b, const float* oc_w, const float* oc_b,
    const float* l1g, const float* l1bb, const float* l1m, const float* l1v,
    const float* l2g, const float* l2bb, const float* l2m, const float* l2v,
    const float* ocg, const float* ocbb, const float* ocm, const float* ocv,
    char* ws)
{
  _Float16* bt1  = (_Float16*)(ws + OFF_BT1);
  float*    b1a  = (float*)(ws + OFF_B1);
  _Float16* bt2  = (_Float16*)(ws + OFF_BT2);
  float*    b2a  = (float*)(ws + OFF_B2);
  _Float16* btq  = (_Float16*)(ws + OFF_BTQKV);
  float*    bq   = (float*)(ws + OFF_BQKV);
  _Float16* btl1 = (_Float16*)(ws + OFF_BTL1);
  _Float16* btl2 = (_Float16*)(ws + OFF_BTL2);
  _Float16* btoc = (_Float16*)(ws + OFF_BTOC);
  float*    fold = (float*)(ws + OFF_FOLD);
  float*    ksum = (float*)(ws + OFF_KSUM);
  float*    vsum = (float*)(ws + OFF_VSUM);
  float*    mat  = (float*)(ws + OFF_MAT);
  _Float16* hyp  = (_Float16*)(ws + OFF_HYP);

  int tid = blockIdx.x * blockDim.x + threadIdx.x;
  int nth = gridDim.x * blockDim.x;
  for (int i = tid; i < 65536; i += nth) {
    int n = i >> 8, c = i & 255;
    int o = n & 63, g = n >> 6;
    bt1[i] = (_Float16)up1_w[((c * 64 + o) * 2 + (g >> 1)) * 2 + (g & 1)];
  }
  for (int i = tid; i < 256; i += nth) b1a[i] = up1_b[i & 63];
  for (int i = tid; i < 8192; i += nth) {
    int n = i >> 6, c = i & 63;
    int o = n & 31, g = n >> 5;
    bt2[i] = (_Float16)up2_w[((c * 32 + o) * 2 + (g >> 1)) * 2 + (g & 1)];
  }
  for (int i = tid; i < 128; i += nth) b2a[i] = up2_b[i & 31];
  for (int i = tid; i < 1536; i += nth) {
    int n = i >> 5, c = i & 31;
    float w = (n < 8) ? q_w[n * 32 + c] : (n < 16) ? k_w[(n - 8) * 32 + c] : v_w[(n - 16) * 32 + c];
    btq[i] = (_Float16)w;
  }
  for (int i = tid; i < 48; i += nth)
    bq[i] = (i < 8) ? q_b[i] : (i < 16) ? k_b[i - 8] : v_b[i - 16];
  for (int i = tid; i < 9216; i += nth) {
    int o = i / 288, rem = i % 288, j = rem >> 5, c = rem & 31;
    btl1[i] = (_Float16)l1_w[((o * 32 + c) * 3 + j / 3) * 3 + j % 3];
  }
  for (int i = tid; i < 1024; i += nth) {
    btl2[i] = (_Float16)l2_w[i];
    btoc[i] = (_Float16)oc_w[i];
  }
  for (int i = tid; i < 32; i += nth) {
    float s1 = l1g[i] * rsqrtf(l1v[i] + 1e-5f);
    fold[i] = s1;        fold[32 + i] = (l1_b[i] - l1m[i]) * s1 + l1bb[i];
    float s2 = l2g[i] * rsqrtf(l2v[i] + 1e-5f);
    fold[64 + i] = s2;   fold[96 + i] = (l2_b[i] - l2m[i]) * s2 + l2bb[i];
    float s3 = ocg[i] * rsqrtf(ocv[i] + 1e-5f);
    fold[128 + i] = s3;  fold[160 + i] = (oc_b[i] - ocm[i]) * s3 + ocbb[i];
  }
  for (int i = tid; i < 128; i += nth) ksum[i] = 0.f;
  for (int i = tid; i < 512; i += nth) vsum[i] = 0.f;
  for (int i = tid; i < 4096; i += nth) mat[i] = 0.f;
  for (int i = tid; i < 16384; i += nth) hyp[i] = (_Float16)0;
}

// ---------------- k1: up1 ConvT (GEMM 65536x256x256) + LN + GELU ------------
__global__ void k1_up1(const float* __restrict__ src, const float* __restrict__ lng,
                       const float* __restrict__ lnb, char* ws)
{
  const _Float16* bt1 = (const _Float16*)(ws + OFF_BT1);
  const float* b1a = (const float*)(ws + OFF_B1);
  _Float16* u16 = (_Float16*)(ws + OFF_U16);
  int lane = threadIdx.x & 31, wvid = threadIdx.x >> 5;
  int strip = blockIdx.x * 4 + wvid;            // 0..4095
  int g = blockIdx.y;                           // kidx group 0..3
  int row0 = strip * 16;
  int colT = lane & 15, hi = (lane >> 4) & 1;

  v8f acc[4];
#pragma unroll
  for (int t = 0; t < 4; ++t) acc[t] = zero8();
  const float* arow = src + (size_t)(row0 + colT) * 256;
#pragma unroll
  for (int kb = 0; kb < 8; ++kb) {
    v16h a = ldA32f(arow + kb * 32, lane);
#pragma unroll
    for (int t = 0; t < 4; ++t)
      acc[t] = wmma16(a, ldB16(bt1, 256, lane, g * 4 + t, kb * 32), acc[t]);
  }
  float bias[4], gam[4], bet[4];
#pragma unroll
  for (int t = 0; t < 4; ++t) {
    int o = t * 16 + colT;
    bias[t] = b1a[g * 64 + o]; gam[t] = lng[o]; bet[t] = lnb[o];
  }
  int kh = g >> 1, kw = g & 1;
#pragma unroll
  for (int r = 0; r < 8; ++r) {
    float v[4], s = 0.f, s2 = 0.f;
#pragma unroll
    for (int t = 0; t < 4; ++t) { v[t] = acc[t][r] + bias[t]; s += v[t]; s2 += v[t] * v[t]; }
    s  += __shfl_xor(s, 1, 32);  s  += __shfl_xor(s, 2, 32);
    s  += __shfl_xor(s, 4, 32);  s  += __shfl_xor(s, 8, 32);
    s2 += __shfl_xor(s2, 1, 32); s2 += __shfl_xor(s2, 2, 32);
    s2 += __shfl_xor(s2, 4, 32); s2 += __shfl_xor(s2, 8, 32);
    float mean = s * (1.f / 64.f);
    float var = s2 * (1.f / 64.f) - mean * mean;
    float rstd = rsqrtf(var + 1e-6f);
    int rr = row0 + r + hi * 8;                 // input pixel row (b*4096 + h*64 + w)
    int b = rr >> 12, hwi = rr & 4095, h = hwi >> 6, w = hwi & 63;
    _Float16* orow = u16 + ((size_t)b * 16384 + (size_t)(2 * h + kh) * 128 + (2 * w + kw)) * 64;
#pragma unroll
    for (int t = 0; t < 4; ++t) {
      float x = (v[t] - mean) * rstd * gam[t] + bet[t];
      orow[t * 16 + colT] = (_Float16)gelu_exact(x);
    }
  }
}

// ---------------- k2: up2 ConvT (GEMM 262144x64x128) + GELU + details -------
__global__ void k2_up2(const float* __restrict__ details, char* ws)
{
  const _Float16* u16 = (const _Float16*)(ws + OFF_U16);
  const _Float16* bt2 = (const _Float16*)(ws + OFF_BT2);
  const float* b2a = (const float*)(ws + OFF_B2);
  _Float16* up0 = (_Float16*)(ws + OFF_UP0);
  int lane = threadIdx.x & 31, wvid = threadIdx.x >> 5;
  int strip = blockIdx.x * 4 + wvid;            // 0..16383
  int row0 = strip * 16;
  int colT = lane & 15, hi = (lane >> 4) & 1;

  v8f acc[8];
#pragma unroll
  for (int t = 0; t < 8; ++t) acc[t] = zero8();
  const _Float16* arow = u16 + (size_t)(row0 + colT) * 64;
#pragma unroll
  for (int kb = 0; kb < 2; ++kb) {
    v16h a = ldA16(arow + kb * 32, lane);
#pragma unroll
    for (int t = 0; t < 8; ++t)
      acc[t] = wmma16(a, ldB16(bt2, 64, lane, t, kb * 32), acc[t]);
  }
#pragma unroll
  for (int r = 0; r < 8; ++r) {
    int rr = row0 + r + hi * 8;                 // b*16384 + hh*128 + ww
    int b = rr >> 14, l = rr & 16383, hh = l >> 7, ww = l & 127;
#pragma unroll
    for (int t = 0; t < 8; ++t) {
      int col = t * 16 + colT, g = col >> 5, o = col & 31;
      float x = gelu_exact(acc[t][r] + b2a[col]);
      int Y = 2 * hh + (g >> 1), X = 2 * ww + (g & 1);
      float det = details[(((size_t)b * 32 + o) * 256 + Y) * 256 + X];
      up0[((size_t)b * 65536 + (size_t)Y * 256 + X) * 32 + o] = (_Float16)(x + det);
    }
  }
}

// ---------------- k3: QKV 1x1 conv + l2norm + ksum/vsum partial reduce ------
__global__ void k3_qkv(char* ws)
{
  const _Float16* up0 = (const _Float16*)(ws + OFF_UP0);
  const _Float16* btq = (const _Float16*)(ws + OFF_BTQKV);
  const float* bq = (const float*)(ws + OFF_BQKV);
  _Float16* qn16 = (_Float16*)(ws + OFF_QN);
  _Float16* kn16 = (_Float16*)(ws + OFF_KN);
  _Float16* v16  = (_Float16*)(ws + OFF_V16);
  float* ksum = (float*)(ws + OFF_KSUM);
  float* vsum = (float*)(ws + OFF_VSUM);
  int lane = threadIdx.x & 31, wvid = threadIdx.x >> 5;
  int b = blockIdx.y;
  int waveId = blockIdx.x * 4 + wvid;           // 0..255 per batch
  int colT = lane & 15, hi = (lane >> 4) & 1;
  const _Float16* base = up0 + (size_t)b * 65536 * 32;

  float kpart = 0.f, vpart0 = 0.f, vpart1 = 0.f;
  for (int it = 0; it < 16; ++it) {
    int row0 = (waveId * 16 + it) * 16;         // local pixel base
    v16h a = ldA16(base + (size_t)(row0 + colT) * 32, lane);
    v8f a0 = zero8(), a1 = zero8(), a2 = zero8();
    a0 = wmma16(a, ldB16(btq, 32, lane, 0, 0), a0);
    a1 = wmma16(a, ldB16(btq, 32, lane, 1, 0), a1);
    a2 = wmma16(a, ldB16(btq, 32, lane, 2, 0), a2);
#pragma unroll
    for (int r = 0; r < 8; ++r) {
      int rowp = row0 + r + hi * 8;
      size_t gp = (size_t)b * 65536 + rowp;
      // tile0: cols 0..7 = q, 8..15 = k -> per-pixel l2 normalize over 8 chans
      float v0 = a0[r] + bq[colT];
      float sq = v0 * v0;
      sq += __shfl_xor(sq, 1, 32); sq += __shfl_xor(sq, 2, 32); sq += __shfl_xor(sq, 4, 32);
      float nv = v0 * rsqrtf(sq);
      if (colT < 8) {
        qn16[gp * 8 + colT] = (_Float16)nv;
      } else {
        kn16[gp * 8 + (colT - 8)] = (_Float16)nv;
        kpart += nv;
      }
      float v1 = a1[r] + bq[16 + colT];
      float v2 = a2[r] + bq[32 + colT];
      v16[gp * 32 + colT] = (_Float16)v1;
      v16[gp * 32 + 16 + colT] = (_Float16)v2;
      vpart0 += v1; vpart1 += v2;
    }
  }
  kpart += __shfl_xor(kpart, 16, 32);
  vpart0 += __shfl_xor(vpart0, 16, 32);
  vpart1 += __shfl_xor(vpart1, 16, 32);
  if (lane >= 8 && lane < 16) atomicAdd(&ksum[b * 8 + (colT - 8)], kpart);
  if (lane < 16) {
    atomicAdd(&vsum[b * 32 + colT], vpart0);
    atomicAdd(&vsum[b * 32 + 16 + colT], vpart1);
  }
}

// ---------------- k3b: mat = Kn^T x V  (WMMA over pixel-K) ------------------
__global__ void k3b_mat(char* ws)
{
  const _Float16* kn16 = (const _Float16*)(ws + OFF_KN);
  const _Float16* v16  = (const _Float16*)(ws + OFF_V16);
  float* mat = (float*)(ws + OFF_MAT);
  int lane = threadIdx.x & 31, wvid = threadIdx.x >> 5;
  int b = blockIdx.y;
  int waveId = blockIdx.x * 4 + wvid;           // 0..63 per batch
  int m = lane & 15;
  int b0 = (lane & 16) ? 8 : 0;
  int kb16 = (lane & 16) ? 16 : 0;
  const _Float16* knb = kn16 + (size_t)b * 65536 * 8;
  const _Float16* vb  = v16 + (size_t)b * 65536 * 32;

  v8f m0 = zero8(), m1 = zero8();
  for (int it = 0; it < 32; ++it) {
    int pix0 = waveId * 1024 + it * 32;
    v16h a;
#pragma unroll
    for (int i = 0; i < 8; ++i) {
      int p0 = pix0 + b0 + i;
      int p1 = pix0 + 16 + b0 + i;
      a[i]     = (m < 8) ? knb[(size_t)p0 * 8 + m] : (_Float16)0;
      a[8 + i] = (m < 8) ? knb[(size_t)p1 * 8 + m] : (_Float16)0;
    }
    v16h bv0, bv1;
#pragma unroll
    for (int i = 0; i < 16; ++i) {
      int p = pix0 + ((i < 8) ? b0 + i : 16 + b0 + (i - 8));
      bv0[i] = vb[(size_t)p * 32 + m];
      bv1[i] = vb[(size_t)p * 32 + 16 + m];
    }
    m0 = wmma16(a, bv0, m0);
    m1 = wmma16(a, bv1, m1);
  }
  if (!(lane & 16)) {
#pragma unroll
    for (int r = 0; r < 8; ++r) {
      atomicAdd(&mat[(b * 8 + r) * 32 + m], m0[r]);
      atomicAdd(&mat[(b * 8 + r) * 32 + 16 + m], m1[r]);
    }
  }
}

// ---------------- k3c: build BTmat (cols 0..31 = mat^T, col 32 = ksum+eps) --
__global__ void k3c_prep(char* ws)
{
  const float* mat = (const float*)(ws + OFF_MAT);
  const float* ksum = (const float*)(ws + OFF_KSUM);
  _Float16* btm = (_Float16*)(ws + OFF_BTMAT);
  int b = blockIdx.x;
  for (int i = threadIdx.x; i < 48 * 32; i += blockDim.x) {
    int n = i >> 5, k = i & 31;
    float val = 0.f;
    if (k < 8) {
      if (n < 32) val = mat[(b * 8 + k) * 32 + n];
      else if (n == 32) val = ksum[b * 8 + k] + 1e-6f;
    }
    btm[((size_t)b * 48 + n) * 32 + k] = (_Float16)val;
  }
}

// ---------------- k4: attention apply (tail) + l2 branch -> part ------------
__global__ void k4_attn(char* ws)
{
  const _Float16* qn16 = (const _Float16*)(ws + OFF_QN);
  const _Float16* up0  = (const _Float16*)(ws + OFF_UP0);
  const _Float16* btm  = (const _Float16*)(ws + OFF_BTMAT);
  const _Float16* btl2 = (const _Float16*)(ws + OFF_BTL2);
  const float* vsum = (const float*)(ws + OFF_VSUM);
  const float* fold = (const float*)(ws + OFF_FOLD);
  _Float16* part = (_Float16*)(ws + OFF_PART);
  int lane = threadIdx.x & 31, wvid = threadIdx.x >> 5;
  int strip = blockIdx.x * 4 + wvid;            // 0..65535 global
  int b = strip >> 12;
  size_t row0 = (size_t)strip * 16;
  int colT = lane & 15, hi = (lane >> 4) & 1;

  // A fragment from qn (K=8 valid, rest zero)
  v16h aq = zero16h();
  if (!(lane & 16)) {
    v8h q = *(const v8h*)(qn16 + (row0 + colT) * 8);
#pragma unroll
    for (int i = 0; i < 8; ++i) aq[i] = q[i];
  }
  const _Float16* bt = btm + (size_t)b * 48 * 32;
  v8f am0 = zero8(), am1 = zero8(), am2 = zero8();
  am0 = wmma16(aq, ldB16(bt, 32, lane, 0, 0), am0);
  am1 = wmma16(aq, ldB16(bt, 32, lane, 1, 0), am1);
  am2 = wmma16(aq, ldB16(bt, 32, lane, 2, 0), am2);
  // l2 branch GEMM
  v16h au = ldA16(up0 + (row0 + colT) * 32, lane);
  v8f al0 = zero8(), al1 = zero8();
  al0 = wmma16(au, ldB16(btl2, 32, lane, 0, 0), al0);
  al1 = wmma16(au, ldB16(btl2, 32, lane, 1, 0), al1);

  float vs0 = vsum[b * 32 + colT], vs1 = vsum[b * 32 + 16 + colT];
  float l2s0 = fold[64 + colT], l2b0 = fold[96 + colT];
  float l2s1 = fold[64 + 16 + colT], l2b1 = fold[96 + 16 + colT];
#pragma unroll
  for (int r = 0; r < 8; ++r) {
    float s = __shfl(am2[r], lane & 16, 32);    // col 32 = qn.(ksum+eps)
    float tail = 1.0f / (65536.0f + s);
    size_t gp = row0 + r + hi * 8;
    _Float16* prow = part + gp * 32;
    float wv0 = (vs0 + am0[r]) * tail;
    float wv1 = (vs1 + am1[r]) * tail;
    float b0v = fmaxf(al0[r] * l2s0 + l2b0, 0.f);
    float b1v = fmaxf(al1[r] * l2s1 + l2b1, 0.f);
    prow[colT] = (_Float16)(wv0 + b0v);
    prow[16 + colT] = (_Float16)(wv1 + b1v);
  }
}

// ---------------- k5: 3x3 conv (K=288) + part + oc 1x1 + residual -----------
__global__ void k5_conv(char* ws)
{
  const _Float16* up0  = (const _Float16*)(ws + OFF_UP0);
  const _Float16* part = (const _Float16*)(ws + OFF_PART);
  const _Float16* btl1 = (const _Float16*)(ws + OFF_BTL1);
  const _Float16* btoc = (const _Float16*)(ws + OFF_BTOC);
  const float* fold = (const float*)(ws + OFF_FOLD);
  _Float16* upf = (_Float16*)(ws + OFF_UPF);
  __shared__ __align__(16) _Float16 lds[4][16][32];
  int lane = threadIdx.x & 31, wvid = threadIdx.x >> 5;
  int strip = blockIdx.x * 4 + wvid;            // 0..65535 global
  int b = strip >> 12, loc = strip & 4095;
  int n0 = loc * 16;
  int y = n0 >> 8, x0 = n0 & 255;
  int colT = lane & 15, hi = (lane >> 4) & 1;
  int px = x0 + colT;
  const _Float16* ub = up0 + (size_t)b * 65536 * 32;

  v8f a0 = zero8(), a1 = zero8();
#pragma unroll
  for (int j = 0; j < 9; ++j) {
    int dy = j / 3 - 1, dx = j % 3 - 1;
    int yy = y + dy, xx = px + dx;
    v16h a = zero16h();
    if (yy >= 0 && yy < 256 && xx >= 0 && xx < 256)
      a = ldA16(ub + ((size_t)yy * 256 + xx) * 32, lane);
    a0 = wmma16(a, ldB16(btl1, 288, lane, 0, j * 32), a0);
    a1 = wmma16(a, ldB16(btl1, 288, lane, 1, j * 32), a1);
  }
  float s0 = fold[colT], c0 = fold[32 + colT];
  float s1 = fold[16 + colT], c1 = fold[32 + 16 + colT];
#pragma unroll
  for (int r = 0; r < 8; ++r) {
    int rp = r + hi * 8;
    size_t gp = (size_t)b * 65536 + n0 + rp;
    float w0 = (float)part[gp * 32 + colT] + fmaxf(a0[r] * s0 + c0, 0.f);
    float w1 = (float)part[gp * 32 + 16 + colT] + fmaxf(a1[r] * s1 + c1, 0.f);
    lds[wvid][rp][colT] = (_Float16)w0;
    lds[wvid][rp][16 + colT] = (_Float16)w1;
  }
  __syncthreads();
  v16h a = ldA16(&lds[wvid][colT][0], lane);
  v8f o0 = zero8(), o1 = zero8();
  o0 = wmma16(a, ldB16(btoc, 32, lane, 0, 0), o0);
  o1 = wmma16(a, ldB16(btoc, 32, lane, 1, 0), o1);
  float os0 = fold[128 + colT], oc0 = fold[160 + colT];
  float os1 = fold[128 + 16 + colT], oc1 = fold[160 + 16 + colT];
#pragma unroll
  for (int r = 0; r < 8; ++r) {
    size_t gp = (size_t)b * 65536 + n0 + r + hi * 8;
    float v0 = fmaxf(o0[r] * os0 + oc0, 0.f) + (float)up0[gp * 32 + colT];
    float v1 = fmaxf(o1[r] * os1 + oc1, 0.f) + (float)up0[gp * 32 + 16 + colT];
    upf[gp * 32 + colT] = (_Float16)v0;
    upf[gp * 32 + 16 + colT] = (_Float16)v1;
  }
}

// ---------------- k6: masks = hyper . up  (GEMM K=32, N=20 pad 32) ----------
__global__ void k6_masks(char* ws, float* __restrict__ out)
{
  const _Float16* upf = (const _Float16*)(ws + OFF_UPF);
  const _Float16* hyp = (const _Float16*)(ws + OFF_HYP);
  int lane = threadIdx.x & 31, wvid = threadIdx.x >> 5;
  int strip = blockIdx.x * 4 + wvid;            // 0..65535
  int b = strip >> 12;
  size_t row0 = (size_t)strip * 16;
  int colT = lane & 15, hi = (lane >> 4) & 1;

  v16h a = ldA16(upf + (row0 + colT) * 32, lane);
  const _Float16* hb = hyp + (size_t)b * 32 * 32;
  v8f c0 = zero8(), c1 = zero8();
  c0 = wmma16(a, ldB16(hb, 32, lane, 0, 0), c0);
  c1 = wmma16(a, ldB16(hb, 32, lane, 1, 0), c1);
  int nloc = (strip & 4095) * 16;
#pragma unroll
  for (int r = 0; r < 8; ++r) {
    int n = nloc + r + hi * 8;
    out[((size_t)b * 20 + colT) * 65536 + n] = c0[r];
    if (16 + colT < 20)
      out[((size_t)b * 20 + 16 + colT) * 65536 + n] = c1[r];
  }
}

// ---------------- hyper MLP + IOU MLP ---------------------------------------
__global__ void k_mlp(const float* __restrict__ mask_tok, const float* __restrict__ iou_tok,
                      const float* hw1, const float* hb1, const float* hw2, const float* hb2,
                      const float* hw3, const float* hb3,
                      const float* iw1, const float* ib1, const float* iw2, const float* ib2,
                      const float* iw3, const float* ib3, const int* msp,
                      char* ws, float* __restrict__ out)
{
  __shared__ float t0[256], h1[256], h2[256];
  _Float16* hyp = (_Float16*)(ws + OFF_HYP);
  int k = blockIdx.x, b = blockIdx.y, h = threadIdx.x;
  bool iou = (k == 20);
  int ms = msp[0];
  t0[h] = iou ? iou_tok[b * 256 + h] : mask_tok[((size_t)(b * 4 + ms)) * 256 + h];
  __syncthreads();
  const float* W1 = iou ? iw1 : hw1 + (size_t)k * 65536;
  const float* B1 = iou ? ib1 : hb1 + k * 256;
  const float* W2 = iou ? iw2 : hw2 + (size_t)k * 65536;
  const float* B2 = iou ? ib2 : hb2 + k * 256;
  float acc = 0.f;
  for (int d = 0; d < 256; ++d) acc += t0[d] * W1[d * 256 + h];
  h1[h] = fmaxf(acc + B1[h], 0.f);
  __syncthreads();
  acc = 0.f;
  for (int d = 0; d < 256; ++d) acc += h1[d] * W2[d * 256 + h];
  h2[h] = fmaxf(acc + B2[h], 0.f);
  __syncthreads();
  if (!iou) {
    if (h < 32) {
      float a3 = 0.f;
      for (int d = 0; d < 256; ++d) a3 += h2[d] * hw3[((size_t)k * 256 + d) * 32 + h];
      hyp[((size_t)b * 32 + k) * 32 + h] = (_Float16)(a3 + hb3[k * 32 + h]);
    }
  } else {
    if (h < 4) {
      float a3 = 0.f;
      for (int d = 0; d < 256; ++d) a3 += h2[d] * iw3[d * 4 + h];
      out[MASK_ELEMS + b * 4 + h] = a3 + ib3[h];
    }
  }
}

// ---------------- launch -----------------------------------------------------
extern "C" void kernel_launch(void* const* d_in, const int* in_sizes, int n_in,
                              void* d_out, int out_size, void* d_ws, size_t ws_size,
                              hipStream_t stream)
{
  (void)in_sizes; (void)n_in; (void)out_size; (void)ws_size;
  const float* src      = (const float*)d_in[0];
  const float* iou_tok  = (const float*)d_in[1];
  const float* mask_tok = (const float*)d_in[2];
  const float* details  = (const float*)d_in[3];
  const float* P[42];
  for (int i = 0; i < 42; ++i) P[i] = (const float*)d_in[4 + i];
  const int* msp = (const int*)d_in[46];
  float* out = (float*)d_out;
  char* ws = (char*)d_ws;

  // P indices: 0 up1_w 1 up1_b 2 ln_g 3 ln_b 4 up2_w 5 up2_b 6 q_w 7 q_b
  // 8 k_w 9 k_b 10 v_w 11 v_b 12 l1_w 13 l1_b 14 l2_w 15 l2_b 16 oc_w 17 oc_b
  // 18 hw1 19 hb1 20 hw2 21 hb2 22 hw3 23 hb3 24 iw1 25 ib1 26 iw2 27 ib2
  // 28 iw3 29 ib3 30..33 l1_g/bb/m/v 34..37 l2_* 38..41 oc_*
  k0_pack<<<64, 256, 0, stream>>>(
      P[0], P[1], P[4], P[5], P[6], P[7], P[8], P[9], P[10], P[11],
      P[12], P[13], P[14], P[15], P[16], P[17],
      P[30], P[31], P[32], P[33], P[34], P[35], P[36], P[37],
      P[38], P[39], P[40], P[41], ws);

  k_mlp<<<dim3(21, 16), 256, 0, stream>>>(
      mask_tok, iou_tok, P[18], P[19], P[20], P[21], P[22], P[23],
      P[24], P[25], P[26], P[27], P[28], P[29], msp, ws, out);

  k1_up1<<<dim3(1024, 4), 128, 0, stream>>>(src, P[2], P[3], ws);
  k2_up2<<<4096, 128, 0, stream>>>(details, ws);
  k3_qkv<<<dim3(64, 16), 128, 0, stream>>>(ws);
  k3b_mat<<<dim3(16, 16), 128, 0, stream>>>(ws);
  k3c_prep<<<16, 256, 0, stream>>>(ws);
  k4_attn<<<16384, 128, 0, stream>>>(ws);
  k5_conv<<<16384, 128, 0, stream>>>(ws);
  k6_masks<<<16384, 128, 0, stream>>>(ws, out);
}
